// TemporalGCN_20134806683790
// MI455X (gfx1250) — compile-verified, matrix operand order checked by reading, not run on hardware
//
#include <hip/hip_runtime.h>
#include <hip/hip_bf16.h>

// TemporalGCN for MI455X (gfx1250, wave32):
//   h  = relu(X @ Wenc + benc)            -- bf16 WMMA, f32 acc, bf16 output
//   t1 = h @ W1 ; o1 = b1 + t1*dis^2 + edge-scatter(t1)   (f32 atomics in L2)
//   t2 = relu(o1) @ W2 ; o2 = b2 + ... + edge-scatter
//   out = segment-mean(o2, batch) over 64 graphs
//
// Weights are pre-packed into the exact WMMA B-fragment register layout
// (bf16, 32 contiguous bytes per lane per K=32 step) so GEMMs load operands
// with b128 vector loads only. Feature tables are 12.8-25.6 MB -> fully
// L2-resident (192 MB); the edge scatter is L2-atomic bound, not HBM bound.

#define GRAPHS 64

typedef __attribute__((ext_vector_type(16))) __bf16 v16bf;
typedef __attribute__((ext_vector_type(8)))  __bf16 v8bf;
typedef __attribute__((ext_vector_type(8)))  float  v8f;

// ---------------------------------------------------------------------------
// Pack W[K x 64] (f32, row-major) into WMMA B fragments (bf16), zero-padded.
// Tile t = ks*4 + nt covers K-rows [ks*32, ks*32+32), cols [nt*16, nt*16+16).
// Fragment element j (0..15) of lane l:
//   col = nt*16 + (l&15),  k = ks*32 + (j<8 ? (l>>4)*8+j : 16+(l>>4)*8+j-8)
// Stored contiguously: Wp[((t*32)+l)*16 + j]  -> one 32B chunk per lane.
// ---------------------------------------------------------------------------
__global__ void k_pack_w(const float* __restrict__ W, __bf16* __restrict__ Wp,
                         int K, int nks) {
    const int tid = blockIdx.x * blockDim.x + threadIdx.x;   // nks*4*32*16
    if (tid >= nks * 4 * 512) return;
    const int j    = tid & 15;
    const int lane = (tid >> 4) & 31;
    const int t    = tid >> 9;
    const int ks   = t >> 2;
    const int nt   = t & 3;
    const int ksel = lane >> 4;
    const int col  = (nt << 4) + (lane & 15);
    const int k    = ks * 32 + ((j < 8) ? (ksel * 8 + j)
                                        : (16 + ksel * 8 + (j - 8)));
    const float v  = (k < K) ? W[(size_t)k * 64 + col] : 0.0f;
    Wp[tid] = (__bf16)v;
}

// ---------------------------------------------------------------------------
// Encoder GEMM: Hbf[M x 64] = relu(X[M x 200] @ Wenc + bias), bf16 out.
// One wave -> one 16x16 tile; block = 8 waves = 32 rows x 64 cols.
// ---------------------------------------------------------------------------
__global__ __launch_bounds__(256)
void gemm_enc(const float* __restrict__ X, const __bf16* __restrict__ Wp,
              const float* __restrict__ bias, __bf16* __restrict__ Hbf, int M)
{
    constexpr int K = 200;                       // KPAD = 224 -> 7 K-steps
    const int wave = threadIdx.x >> 5;
    const int lane = threadIdx.x & 31;
    const int m0   = (blockIdx.x * 2 + (wave >> 2)) * 16;
    const int nt   = wave & 3;
    if (m0 >= M) return;

    const int hl   = lane & 15;
    const int ksel = lane >> 4;
    const float* ar = X + (size_t)(m0 + hl) * K;

    v8f acc = {};
#pragma unroll
    for (int ks = 0; ks < 7; ++ks) {
        const int ka = ks * 32 + (ksel << 3);
        const int kb = ka + 16;
        v16bf afrag;
#pragma unroll
        for (int i = 0; i < 8; ++i) {
            const int k1 = ka + i, k2 = kb + i;
            afrag[i]     = (__bf16)((k1 < K) ? ar[k1] : 0.0f);
            afrag[i + 8] = (__bf16)((k2 < K) ? ar[k2] : 0.0f);
        }
        const v16bf bfrag =
            *(const v16bf*)(Wp + ((size_t)((ks * 4 + nt) * 32 + lane) << 4));
        acc = __builtin_amdgcn_wmma_f32_16x16x32_bf16(
            false, afrag, false, bfrag, (short)0, acc, false, false);
    }

    const int bcol  = (nt << 4) + hl;
    const int rbase = m0 + (ksel << 3);          // C/D: M = m0 + 8*ksel + r
    const float bv  = bias[bcol];
#pragma unroll
    for (int r = 0; r < 8; ++r)
        Hbf[(size_t)(rbase + r) * 64 + bcol] = (__bf16)fmaxf(acc[r] + bv, 0.0f);
}

// ---------------------------------------------------------------------------
// Hidden GEMM: out[M x 64] = Abf[M x 64] @ W (packed), f32 out, no bias.
// A fragment: two b128 loads + shuffle; B fragment: one v16bf load.
// ---------------------------------------------------------------------------
__global__ __launch_bounds__(256)
void gemm_h64(const __bf16* __restrict__ Abf, const __bf16* __restrict__ Wp,
              float* __restrict__ out, int M)
{
    const int wave = threadIdx.x >> 5;
    const int lane = threadIdx.x & 31;
    const int m0   = (blockIdx.x * 2 + (wave >> 2)) * 16;
    const int nt   = wave & 3;
    if (m0 >= M) return;

    const int hl   = lane & 15;
    const int ksel = lane >> 4;
    const __bf16* ap = Abf + (size_t)(m0 + hl) * 64 + (ksel << 3);

    v8f acc = {};
#pragma unroll
    for (int ks = 0; ks < 2; ++ks) {
        const v8bf lo = *(const v8bf*)(ap + ks * 32);
        const v8bf hi = *(const v8bf*)(ap + ks * 32 + 16);
        const v16bf afrag = __builtin_shufflevector(
            lo, hi, 0, 1, 2, 3, 4, 5, 6, 7, 8, 9, 10, 11, 12, 13, 14, 15);
        const v16bf bfrag =
            *(const v16bf*)(Wp + ((size_t)((ks * 4 + nt) * 32 + lane) << 4));
        acc = __builtin_amdgcn_wmma_f32_16x16x32_bf16(
            false, afrag, false, bfrag, (short)0, acc, false, false);
    }

    const int bcol  = (nt << 4) + hl;
    const int rbase = m0 + (ksel << 3);
#pragma unroll
    for (int r = 0; r < 8; ++r)
        out[(size_t)(rbase + r) * 64 + bcol] = acc[r];
}

// ---------------------------------------------------------------------------
// Elementwise / scatter helpers
// ---------------------------------------------------------------------------
__global__ void k_fill(float* __restrict__ p, float v, long long n) {
    long long i = (long long)blockIdx.x * blockDim.x + threadIdx.x;
    if (i < n) p[i] = v;
}

__global__ void k_deg_count(const int* __restrict__ dst, float* __restrict__ deg,
                            long long E) {
    long long e = (long long)blockIdx.x * blockDim.x + threadIdx.x;
    if (e < E) atomicAdd(&deg[dst[e]], 1.0f);
}

__global__ void k_rsqrt(float* __restrict__ d, int n) {
    int i = blockIdx.x * blockDim.x + threadIdx.x;
    if (i < n) d[i] = rsqrtf(d[i]);              // deg >= 1 (self-loop)
}

// out[n][c] = bias[c] + t[n][c] * dis[n]^2  (self-loop term; also inits out)
__global__ void k_selfbias(const float* __restrict__ t, const float* __restrict__ dis,
                           const float* __restrict__ bias, float* __restrict__ out,
                           long long n64) {
    long long i = (long long)blockIdx.x * blockDim.x + threadIdx.x;
    if (i >= n64) return;
    const int node = (int)(i >> 6);
    const float d  = dis[node];
    out[i] = bias[(int)(i & 63)] + t[i] * d * d;
}

// relu + f32 -> bf16 repack (feeds conv2's WMMA A operand)
__global__ void k_relubf(const float* __restrict__ in, __bf16* __restrict__ out,
                         long long n64) {
    long long i = (long long)blockIdx.x * blockDim.x + threadIdx.x;
    if (i < n64) out[i] = (__bf16)fmaxf(in[i], 0.0f);
}

// 16 lanes per edge, float4 per lane: gather t[src], scale, atomic-add out[dst]
__global__ void k_edge_scatter(const float* __restrict__ t, const float* __restrict__ dis,
                               const int* __restrict__ src, const int* __restrict__ dst,
                               float* __restrict__ out, long long E) {
    long long tid = (long long)blockIdx.x * blockDim.x + threadIdx.x;
    long long e   = tid >> 4;
    if (e >= E) return;
    const int c4 = (int)(tid & 15) << 2;
    const int s  = src[e];
    const int d  = dst[e];
    const float nm = dis[s] * dis[d];
    const float4 v = *(const float4*)(t + (size_t)s * 64 + c4);
    float* o = out + (size_t)d * 64 + c4;
    atomicAdd(o + 0, v.x * nm);
    atomicAdd(o + 1, v.y * nm);
    atomicAdd(o + 2, v.z * nm);
    atomicAdd(o + 3, v.w * nm);
}

__global__ void k_pool_accum(const float* __restrict__ h, const int* __restrict__ batch,
                             float* __restrict__ sums, float* __restrict__ counts,
                             long long n16) {
    long long tid = (long long)blockIdx.x * blockDim.x + threadIdx.x;
    if (tid >= n16) return;
    const int node = (int)(tid >> 4);
    const int c4   = (int)(tid & 15) << 2;
    const int g    = batch[node];
    const float4 v = *(const float4*)(h + (size_t)node * 64 + c4);
    float* s = sums + (size_t)g * 64 + c4;
    atomicAdd(s + 0, v.x);
    atomicAdd(s + 1, v.y);
    atomicAdd(s + 2, v.z);
    atomicAdd(s + 3, v.w);
    if ((tid & 15) == 0) atomicAdd(&counts[g], 1.0f);
}

__global__ void k_pool_final(const float* __restrict__ sums,
                             const float* __restrict__ counts,
                             float* __restrict__ out) {
    int i = blockIdx.x * blockDim.x + threadIdx.x;
    if (i < GRAPHS * 64) out[i] = sums[i] / fmaxf(counts[i >> 6], 1.0f);
}

// ---------------------------------------------------------------------------
// Launch
// ---------------------------------------------------------------------------
extern "C" void kernel_launch(void* const* d_in, const int* in_sizes, int n_in,
                              void* d_out, int out_size, void* d_ws, size_t ws_size,
                              hipStream_t stream) {
    const float* x     = (const float*)d_in[0];   // [N, 200]
    const int*   eidx  = (const int*)d_in[1];     // [2, E] (JAX default int32)
    const int*   batch = (const int*)d_in[2];     // [N]
    const float* Wenc  = (const float*)d_in[3];   // [200, 64]
    const float* benc  = (const float*)d_in[4];
    const float* W1    = (const float*)d_in[5];   // [64, 64]
    const float* b1    = (const float*)d_in[6];
    const float* W2    = (const float*)d_in[7];
    const float* b2    = (const float*)d_in[8];
    float*       out   = (float*)d_out;           // [64, 64]

    const int       N = in_sizes[0] / 200;        // 100000 (multiple of 32)
    const long long E = (long long)in_sizes[1] / 2;
    const int* src = eidx;
    const int* dst = eidx + E;

    // workspace (float granularity, every offset 64B-aligned):
    float* ws     = (float*)d_ws;
    float*  dis   = ws;                                   // N
    __bf16* hencb = (__bf16*)(dis + N);                   // N*64 bf16
    __bf16* abf2  = (__bf16*)(dis + N + (size_t)N * 32);  // N*64 bf16
    float*  tbuf  = dis + N + (size_t)N * 64;             // N*64 f32
    float*  o1    = tbuf + (size_t)N * 64;                // N*64 f32
    float*  o2    = o1;                                   // o1 dead after relubf
    float*  sums  = o1 + (size_t)N * 64;                  // G*64
    float*  cnts  = sums + GRAPHS * 64;                   // G
    __bf16* wpE   = (__bf16*)(cnts + 64);                 // 7*4*512 bf16
    __bf16* wp1   = wpE + 7 * 4 * 512;                    // 2*4*512 bf16
    __bf16* wp2   = wp1 + 2 * 4 * 512;

    const long long n64 = (long long)N * 64;
    dim3 blk(256);

    // degrees with self-loops -> dis = rsqrt(deg)
    k_fill     <<<(N + 255) / 256, blk, 0, stream>>>(dis, 1.0f, N);
    k_deg_count<<<(int)((E + 255) / 256), blk, 0, stream>>>(dst, dis, E);
    k_rsqrt    <<<(N + 255) / 256, blk, 0, stream>>>(dis, N);

    // pack weights into WMMA B-fragment layout (bf16)
    k_pack_w<<<(7 * 4 * 512 + 255) / 256, blk, 0, stream>>>(Wenc, wpE, 200, 7);
    k_pack_w<<<(2 * 4 * 512 + 255) / 256, blk, 0, stream>>>(W1,   wp1,  64, 2);
    k_pack_w<<<(2 * 4 * 512 + 255) / 256, blk, 0, stream>>>(W2,   wp2,  64, 2);

    // encoder: hencb = relu(x @ Wenc + benc)   (bf16 output)
    gemm_enc<<<(N + 31) / 32, blk, 0, stream>>>(x, wpE, benc, hencb, N);

    // conv1: t1 = hencb @ W1 ; o1 = b1 + t1*dis^2 + scatter(t1)
    gemm_h64      <<<(N + 31) / 32, blk, 0, stream>>>(hencb, wp1, tbuf, N);
    k_selfbias    <<<(int)((n64 + 255) / 256), blk, 0, stream>>>(tbuf, dis, b1, o1, n64);
    k_edge_scatter<<<(int)((E * 16 + 255) / 256), blk, 0, stream>>>(tbuf, dis, src, dst, o1, E);

    // conv2: abf2 = bf16(relu(o1)) ; t2 = abf2 @ W2 ; o2 = b2 + ... + scatter
    k_relubf      <<<(int)((n64 + 255) / 256), blk, 0, stream>>>(o1, abf2, n64);
    gemm_h64      <<<(N + 31) / 32, blk, 0, stream>>>(abf2, wp2, tbuf, N);
    k_selfbias    <<<(int)((n64 + 255) / 256), blk, 0, stream>>>(tbuf, dis, b2, o2, n64);
    k_edge_scatter<<<(int)((E * 16 + 255) / 256), blk, 0, stream>>>(tbuf, dis, src, dst, o2, E);

    // global mean pool
    k_fill      <<<(GRAPHS * 65 + 255) / 256, blk, 0, stream>>>(sums, 0.0f, GRAPHS * 65);
    k_pool_accum<<<(int)(((long long)N * 16 + 255) / 256), blk, 0, stream>>>(o2, batch, sums, cnts, (long long)N * 16);
    k_pool_final<<<(GRAPHS * 64 + 255) / 256, blk, 0, stream>>>(sums, cnts, out);
}
